// MultiHeadAttention_11467562680711
// MI455X (gfx1250) — compile-verified, matrix operand order checked by reading, not run on hardware
//
#include <hip/hip_runtime.h>

#define BATCH 4
#define NSEQ  2048
#define EMB   1024
#define HEADS 16
#define HD    64

typedef __attribute__((ext_vector_type(16))) __bf16 v16bf;
typedef __attribute__((ext_vector_type(8)))  __bf16 bf16x8;
typedef __attribute__((ext_vector_type(8)))  float  v8f;

// Build a 16-element A/B fragment from two contiguous 8-element (16B) chunks:
// elements 0..7 = p[0..7], 8..15 = p[16..23] (matches lane's K chunks).
__device__ __forceinline__ v16bf load_frag(const __bf16* p) {
  bf16x8 lo = *(const bf16x8*)p;
  bf16x8 hi = *(const bf16x8*)(p + 16);
  v16bf r;
#pragma unroll
  for (int i = 0; i < 8; ++i) { r[i] = lo[i]; r[i + 8] = hi[i]; }
  return r;
}

__device__ __forceinline__ v16bf mk_frag(bf16x8 lo, bf16x8 hi) {
  v16bf r;
#pragma unroll
  for (int i = 0; i < 8; ++i) { r[i] = lo[i]; r[i + 8] = hi[i]; }
  return r;
}

// gfx1250 async copy: 16B per lane, global -> LDS, tracked by ASYNCcnt.
__device__ __forceinline__ void async_b128(unsigned lds_off, const void* gaddr) {
  asm volatile("global_load_async_to_lds_b128 %0, %1, off"
               :: "v"(lds_off), "v"(gaddr) : "memory");
}
__device__ __forceinline__ void wait_asynccnt0() {
  asm volatile("s_wait_asynccnt 0x0" ::: "memory");
}
__device__ __forceinline__ unsigned lds_addr32(const void* p) {
  return (unsigned)(unsigned long long)p;   // flat LDS aperture: addr[31:0] = LDS offset
}

// ---------------- elementwise f32 -> bf16 ----------------
__global__ void k_cvt_bf16(const float* __restrict__ src, __bf16* __restrict__ dst, int n) {
  int i = blockIdx.x * blockDim.x + threadIdx.x;
  int stride = gridDim.x * blockDim.x;
  for (; i < n; i += stride) dst[i] = (__bf16)src[i];
}

// ---------------- transpose+convert weights: dst[o][k] = src[k][o] ----------------
__global__ void k_transpose_bf16(const float* __restrict__ src, __bf16* __restrict__ dst) {
  int k = blockIdx.x * 16 + threadIdx.x;
  int o = blockIdx.y * 16 + threadIdx.y;
  dst[(size_t)o * EMB + k] = (__bf16)src[(size_t)k * EMB + o];
}

// ---------------- WMMA GEMM: Y = X(bf16,row) * W(via wT bf16,[o][k]) + bias ----------------
// Per wave: 32(M) x 64(N) tile; block = 4 waves -> 128(M) x 64(N).
// MODE 0: write bf16 (b,h,n,d);  MODE 1: write bf16 (b,h,d,n);  MODE 2: write f32 (b,n,e)
template <int MODE>
__global__ void k_proj(const __bf16* __restrict__ xb, const __bf16* __restrict__ wT,
                       const float* __restrict__ bias, void* __restrict__ out) {
  const int lane = threadIdx.x & 31;
  const int wave = threadIdx.x >> 5;
  const int ln   = lane & 15;
  const int hif  = lane >> 4;        // 0: K chunks {0-7,16-23}; 1: {8-15,24-31}
  const int koff = hif * 8;
  const int b    = blockIdx.z;
  const int rowBase = blockIdx.x * 128 + wave * 32;
  const int colBase = blockIdx.y * 64;

  const __bf16* arow0 = xb + ((size_t)(b * NSEQ + rowBase + ln)) * EMB + koff;
  const __bf16* arow1 = arow0 + (size_t)16 * EMB;

  v8f acc[2][4] = {};
  for (int k0 = 0; k0 < EMB; k0 += 32) {
    v16bf a0 = load_frag(arow0 + k0);
    v16bf a1 = load_frag(arow1 + k0);
#pragma unroll
    for (int t = 0; t < 4; ++t) {
      const __bf16* brow = wT + ((size_t)(colBase + t * 16 + ln)) * EMB + k0 + koff;
      v16bf bb = load_frag(brow);
      acc[0][t] = __builtin_amdgcn_wmma_f32_16x16x32_bf16(false, a0, false, bb,
                                                          (short)0, acc[0][t], false, false);
      acc[1][t] = __builtin_amdgcn_wmma_f32_16x16x32_bf16(false, a1, false, bb,
                                                          (short)0, acc[1][t], false, false);
    }
  }

#pragma unroll
  for (int m = 0; m < 2; ++m) {
#pragma unroll
    for (int t = 0; t < 4; ++t) {
      const int c = colBase + t * 16 + ln;
      const float bv = bias[c];
#pragma unroll
      for (int v = 0; v < 8; ++v) {
        const int r = rowBase + m * 16 + v + 8 * hif;
        const float val = acc[m][t][v] + bv;
        if (MODE == 0) {
          int h = c >> 6, d = c & 63;
          ((__bf16*)out)[(((size_t)(b * HEADS + h)) * NSEQ + r) * HD + d] = (__bf16)val;
        } else if (MODE == 1) {
          int h = c >> 6, d = c & 63;
          ((__bf16*)out)[(((size_t)(b * HEADS + h)) * HD + d) * NSEQ + r] = (__bf16)val;
        } else {
          ((float*)out)[((size_t)(b * NSEQ + r)) * EMB + c] = val;
        }
      }
    }
  }
}

// ---------------- flash attention ----------------
// Block = 4 waves, each owns a 16-query tile of the same (b,h).
// K/V blocks (32 keys) are double-buffered in LDS via gfx1250 async copies so
// the DMA for block i+1 overlaps the WMMA/softmax work on block i.
__global__ void k_flash(const __bf16* __restrict__ Q, const __bf16* __restrict__ K,
                        const __bf16* __restrict__ Vt, __bf16* __restrict__ attn) {
  __shared__ __bf16 kBuf[2][32][HD];  // 2 x 4 KB: 32 keys x 64 dims (row-major)
  __shared__ __bf16 vBuf[2][HD][32];  // 2 x 4 KB: 64 dims x 32 keys (V^T layout)
  __shared__ float  sS[4][16][32];    // per-wave score tile (C-layout -> row-major)
  __shared__ __bf16 sP[4][16][32];    // per-wave probabilities, bf16, row-major
  __shared__ float  sCorr[4][16];
  __shared__ float  sL[4][16];

  const int tid  = threadIdx.x;
  const int lane = tid & 31;
  const int wave = tid >> 5;
  const int ln   = lane & 15;
  const int hif  = lane >> 4;
  const int koff = hif * 8;
  const int h = blockIdx.y, b = blockIdx.z;
  const int qBase = (blockIdx.x * 4 + wave) * 16;

  const __bf16* Qp = Q  + ((size_t)(b * HEADS + h)) * NSEQ * HD;
  const __bf16* Kp = K  + ((size_t)(b * HEADS + h)) * NSEQ * HD;
  const __bf16* Vp = Vt + ((size_t)(b * HEADS + h)) * HD * NSEQ;

  const unsigned kBase = lds_addr32(&kBuf[0][0][0]);
  const unsigned vBase = lds_addr32(&vBuf[0][0][0]);

  // issue the 4 async-copy instructions for key-block `kb` into buffer `buf`
  auto stage = [&](int buf, int kb) {
    // K: rows kb..kb+31 are contiguous in memory: 256 x 16B chunks, 2/thread
    const __bf16* ksrc = Kp + (size_t)kb * HD;
    const unsigned kb32 = kBase + buf * 4096;
    async_b128(kb32 + tid * 16,        ksrc + tid * 8);
    async_b128(kb32 + 2048 + tid * 16, ksrc + 1024 + tid * 8);
    // V^T: 64 rows (dims) x 32 keys; 4 x 16B chunks per row, 2 chunks/thread
    const unsigned vb32 = vBase + buf * 4096;
#pragma unroll
    for (int cc = 0; cc < 2; ++cc) {
      const int c = tid + cc * 128;            // chunk id 0..255
      const int vr = c >> 2, vc = (c & 3) * 8;
      async_b128(vb32 + c * 16, Vp + (size_t)vr * NSEQ + kb + vc);
    }
  };

  const __bf16* qrow = Qp + (size_t)(qBase + ln) * HD + koff;
  v16bf a0 = load_frag(qrow);        // d 0..31
  v16bf a1 = load_frag(qrow + 32);   // d 32..63

  float m_i = -1e30f, l_i = 0.f;     // only meaningful in hif==0 lanes
  v8f o[4] = {};
  const float scale = 0.03125f;      // 1/sqrt(EMB) = 1/32

  stage(0, 0);                       // prologue: fill buffer 0

  for (int it = 0; it < NSEQ / 32; ++it) {
    const int kb  = it * 32;
    const int cur = it & 1;

    wait_asynccnt0();                // this wave's copies for `cur` done
    __syncthreads();                 // ... and every other wave's too

    if (kb + 32 < NSEQ) stage(cur ^ 1, kb + 32);   // DMA next block during compute

    // ---- S = Q * K^T from LDS (two 16x16 key subtiles, d in 2 chunks) ----
    v8f s0 = {}, s1 = {};
    {
      v16bf b00 = load_frag(&kBuf[cur][ln][koff]);
      v16bf b01 = load_frag(&kBuf[cur][ln][koff + 32]);
      v16bf b10 = load_frag(&kBuf[cur][16 + ln][koff]);
      v16bf b11 = load_frag(&kBuf[cur][16 + ln][koff + 32]);
      s0 = __builtin_amdgcn_wmma_f32_16x16x32_bf16(false, a0, false, b00, (short)0, s0, false, false);
      s0 = __builtin_amdgcn_wmma_f32_16x16x32_bf16(false, a1, false, b01, (short)0, s0, false, false);
      s1 = __builtin_amdgcn_wmma_f32_16x16x32_bf16(false, a0, false, b10, (short)0, s1, false, false);
      s1 = __builtin_amdgcn_wmma_f32_16x16x32_bf16(false, a1, false, b11, (short)0, s1, false, false);
    }

    // ---- C-layout -> row-major in per-wave LDS (DS ops in-order per wave) ----
#pragma unroll
    for (int v = 0; v < 8; ++v) {
      sS[wave][v + 8 * hif][ln]      = s0[v] * scale;
      sS[wave][v + 8 * hif][ln + 16] = s1[v] * scale;
    }
    asm volatile("" ::: "memory");

    // ---- online softmax (row r = ln); low half-lanes only, broadcast via LDS ----
    if (hif == 0) {
      float srow[32];
#pragma unroll
      for (int c = 0; c < 32; ++c) srow[c] = sS[wave][ln][c];
      float mx = m_i;
#pragma unroll
      for (int c = 0; c < 32; ++c) mx = fmaxf(mx, srow[c]);
      const float corr = __expf(m_i - mx);
      float sum = 0.f;
      bf16x8 pb[4];
#pragma unroll
      for (int c = 0; c < 32; ++c) {
        float p = __expf(srow[c] - mx);
        sum += p;
        pb[c >> 3][c & 7] = (__bf16)p;
      }
      l_i = l_i * corr + sum;
      m_i = mx;
#pragma unroll
      for (int c8 = 0; c8 < 4; ++c8)
        *(bf16x8*)&sP[wave][ln][c8 * 8] = pb[c8];
      sCorr[wave][ln] = corr;
    }
    asm volatile("" ::: "memory");

    // ---- P A-fragment: this lane owns cols {koff..koff+7, koff+16..koff+23} ----
    v16bf ap = mk_frag(*(const bf16x8*)&sP[wave][ln][koff],
                       *(const bf16x8*)&sP[wave][ln][koff + 16]);

    // ---- rescale O accumulators: row of element v is (v + 8*hif) ----
    float cv[8];
#pragma unroll
    for (int v = 0; v < 8; ++v) cv[v] = sCorr[wave][v + 8 * hif];
#pragma unroll
    for (int t = 0; t < 4; ++t)
#pragma unroll
      for (int v = 0; v < 8; ++v) o[t][v] *= cv[v];

    // ---- O += P * V from LDS ----
#pragma unroll
    for (int t = 0; t < 4; ++t) {
      v16bf bv = load_frag(&vBuf[cur][t * 16 + ln][koff]);
      o[t] = __builtin_amdgcn_wmma_f32_16x16x32_bf16(false, ap, false, bv, (short)0, o[t], false, false);
    }
    __syncthreads();   // all waves done with buffer `cur` before it is re-staged
  }

  if (hif == 0) sL[wave][ln] = l_i;
  asm volatile("" ::: "memory");
  float lv[8];
#pragma unroll
  for (int v = 0; v < 8; ++v) lv[v] = 1.f / sL[wave][v + 8 * hif];

  // write attn (b, n, e) bf16 for the final projection
#pragma unroll
  for (int t = 0; t < 4; ++t) {
    const int e = h * HD + t * 16 + ln;
#pragma unroll
    for (int v = 0; v < 8; ++v) {
      const int q = qBase + v + 8 * hif;
      attn[((size_t)(b * NSEQ + q)) * EMB + e] = (__bf16)(o[t][v] * lv[v]);
    }
  }
}

extern "C" void kernel_launch(void* const* d_in, const int* in_sizes, int n_in,
                              void* d_out, int out_size, void* d_ws, size_t ws_size,
                              hipStream_t stream) {
  const float* x  = (const float*)d_in[0];
  const float* Wq = (const float*)d_in[1];
  const float* bq = (const float*)d_in[2];
  const float* Wk = (const float*)d_in[3];
  const float* bk = (const float*)d_in[4];
  const float* Wv = (const float*)d_in[5];
  const float* bv = (const float*)d_in[6];
  const float* Wo = (const float*)d_in[7];
  const float* bo = (const float*)d_in[8];
  float* out = (float*)d_out;

  char* ws = (char*)d_ws;
  size_t off = 0;
  __bf16* xb   = (__bf16*)(ws + off); off += (size_t)BATCH * NSEQ * EMB * 2;   // 16 MB
  __bf16* wqT  = (__bf16*)(ws + off); off += (size_t)EMB * EMB * 2;            //  2 MB
  __bf16* wkT  = (__bf16*)(ws + off); off += (size_t)EMB * EMB * 2;
  __bf16* wvT  = (__bf16*)(ws + off); off += (size_t)EMB * EMB * 2;
  __bf16* woT  = (__bf16*)(ws + off); off += (size_t)EMB * EMB * 2;
  __bf16* Qb   = (__bf16*)(ws + off); off += (size_t)BATCH * HEADS * NSEQ * HD * 2; // 16 MB
  __bf16* Kb   = (__bf16*)(ws + off); off += (size_t)BATCH * HEADS * NSEQ * HD * 2;
  __bf16* Vtb  = (__bf16*)(ws + off); off += (size_t)BATCH * HEADS * HD * NSEQ * 2;
  __bf16* attn = (__bf16*)(ws + off); off += (size_t)BATCH * NSEQ * EMB * 2;
  (void)ws_size; (void)in_sizes; (void)n_in; (void)out_size;

  // 1) precision/layout prep
  k_cvt_bf16<<<4096, 256, 0, stream>>>(x, xb, BATCH * NSEQ * EMB);
  dim3 tgrid(EMB / 16, EMB / 16);
  dim3 tblk(16, 16);
  k_transpose_bf16<<<tgrid, tblk, 0, stream>>>(Wq, wqT);
  k_transpose_bf16<<<tgrid, tblk, 0, stream>>>(Wk, wkT);
  k_transpose_bf16<<<tgrid, tblk, 0, stream>>>(Wv, wvT);
  k_transpose_bf16<<<tgrid, tblk, 0, stream>>>(Wo, woT);

  // 2) QKV projections (WMMA)
  dim3 pgrid(NSEQ / 128, EMB / 64, BATCH);
  k_proj<0><<<pgrid, 128, 0, stream>>>(xb, wqT, bq, (void*)Qb);
  k_proj<0><<<pgrid, 128, 0, stream>>>(xb, wkT, bk, (void*)Kb);
  k_proj<1><<<pgrid, 128, 0, stream>>>(xb, wvT, bv, (void*)Vtb);

  // 3) flash attention (WMMA + double-buffered async LDS staging)
  dim3 fgrid(NSEQ / 64, HEADS, BATCH);
  k_flash<<<fgrid, 128, 0, stream>>>(Qb, Kb, Vtb, attn);

  // 4) output projection (WMMA, f32 out)
  k_proj<2><<<pgrid, 128, 0, stream>>>(attn, woT, bo, (void*)out);
}
